// DSSA_18932215840903
// MI455X (gfx1250) — matrix operand order repaired and checked by reading, not compiled, not run on hardware
//
#include <hip/hip_runtime.h>
#include <hip/hip_bf16.h>

// ---------------------------------------------------------------------------
// DSSA spiking-attention block for gfx1250 (MI455X).
// - All GEMMs via v_wmma_f32_16x16x32_f16 (f32 accumulate).
// - Every B operand is stored COLUMN-MAJOR ([n][k]) by its producer, so
//   fragments are contiguous 16-B loads per lane.
// - Large-K GEMMs (k3 patch-embed, k6 projection) stage B tiles into LDS with
//   the Tensor Data Mover, DOUBLE-BUFFERED: issue DMA for chunk k+32, wait
//   tensorcnt<=1 (in-order retirement) so the current tile is resident while
//   the next is in flight. D# pad_enable (16 DW interval + 8 DW pad) yields a
//   48-half padded LDS row stride for conflict-free ds_load_b128 reads.
// - LIF recurrences (T=4) unrolled in registers and fused into the GEMMs.
// ---------------------------------------------------------------------------

typedef __attribute__((ext_vector_type(16))) _Float16 v16h;
typedef __attribute__((ext_vector_type(8)))  _Float16 v8h;
typedef __attribute__((ext_vector_type(8)))  float    v8f;
typedef __attribute__((ext_vector_type(4)))  unsigned v4u;
typedef __attribute__((ext_vector_type(8)))  int      v8i;
typedef __attribute__((ext_vector_type(4)))  int      v4i;

static constexpr int   T_  = 4, B_ = 8, C_ = 384, HW_ = 1024;
static constexpr int   NH  = 12, CH = 32, HP = 8, NPATCH = 64;
static constexpr int   C2  = 768, KC = 6144;          // conv-as-GEMM: M=768, K=C*4*4
static constexpr int   LSTRIDE = 48;                  // padded LDS row (halfs)
static constexpr float TAU_INV = 0.5f;                // TAU = 2.0
static constexpr float EPSV    = 1e-5f;

#if defined(__has_builtin)
#if __has_builtin(__builtin_amdgcn_tensor_load_to_lds)
#define HAVE_TDM 1
#endif
#endif
#ifndef HAVE_TDM
#define HAVE_TDM 0
#endif

__device__ __forceinline__ v8f wmma16x16x32(v16h a, v16h b, v8f c) {
  return __builtin_amdgcn_wmma_f32_16x16x32_f16(false, a, false, b, (short)0, c,
                                                false, false);
}

// A fragment: 16(M) x 32(K) f16, row-major source, leading dim `ld` halfs.
// lane L: M = L&15, Kbase = (L>=16)?8:0 -> two contiguous 16-byte loads.
__device__ __forceinline__ v16h load_a_frag(const _Float16* __restrict__ A, int ld) {
  const int lane = threadIdx.x & 31;
  const int m  = lane & 15;
  const int kb = (lane >> 4) << 3;            // 0 or 8
  const _Float16* p = A + (size_t)m * ld + kb;
  v8h lo = *(const v8h*)p;                    // K = kb .. kb+7
  v8h hi = *(const v8h*)(p + 16);             // K = kb+16 .. kb+23
  v16h r;
#pragma unroll
  for (int i = 0; i < 8; ++i) { r[i] = lo[i]; r[8 + i] = hi[i]; }
  return r;
}

// B fragment from a TRANSPOSED (column-major, [n][k]) source (global or LDS):
// lane L: N = L&15, K = kbase + ((L>=16)?8:0) + {0..7, 16..23} -> two 16-byte loads.
__device__ __forceinline__ v16h load_bt_frag(const _Float16* __restrict__ Bt,
                                             int ld, int kbase) {
  const int lane = threadIdx.x & 31;
  const int n  = lane & 15;
  const int kb = kbase + ((lane >> 4) << 3);  // kbase + 0/8
  const _Float16* p = Bt + (size_t)n * ld + kb;
  v8h lo = *(const v8h*)p;
  v8h hi = *(const v8h*)(p + 16);
  v16h r;
#pragma unroll
  for (int i = 0; i < 8; ++i) { r[i] = lo[i]; r[8 + i] = hi[i]; }
  return r;
}

// Issue the stage of a 64(n) x 32(k) tile from a column-major global matrix
// (row stride `ld` halfs, rows contiguous in k) into LDS rows of 48 halfs.
// TDM path: one tensor_load_to_lds issued by wave 0 (no wait here).
// Fallback: cooperative b128 row copies (visibility via __syncthreads).
__device__ __forceinline__ void stage_issue(const _Float16* __restrict__ g, int ld,
                                            _Float16* __restrict__ lds) {
#if HAVE_TDM
  if ((threadIdx.x >> 5) == 0) {
    const unsigned long long ga = (unsigned long long)(size_t)g;
    v4u g0;
    g0[0] = 1u;                                           // count=1 valid D#
    g0[1] = (unsigned)(size_t)lds;                        // LDS dest (bytes)
    g0[2] = (unsigned)ga;                                 // global_addr[31:0]
    g0[3] = (unsigned)((ga >> 32) & 0x01FFFFFFu) | (2u << 30); // [56:32] | type=2
    v8i g1;
    g1[0] = (1 << 16)   // data_size = 2 bytes
          | (1 << 20)   // pad_enable
          | (3 << 22)   // pad_interval: every 16 DWORDs (= one 32-half row)
          | (7 << 25);  // pad_amount: 8 DWORDs (-> 48-half row stride)
    g1[1] = (int)(((unsigned)ld & 0xFFFFu) << 16);        // tensor_dim0 lo16
    g1[2] = (int)((((unsigned)ld >> 16) & 0xFFFFu) | (64u << 16)); // dim0 hi | dim1=64
    g1[3] = (int)(32u << 16);                             // dim1 hi=0 | tile_dim0=32
    g1[4] = 64;                                           // tile_dim1=64
    g1[5] = ld;                                           // tensor_dim0_stride lo32
    g1[6] = 0;
    g1[7] = 0;
    v4i z4 = {0, 0, 0, 0};
#if __clang_major__ >= 23
    v8i z8 = {0, 0, 0, 0, 0, 0, 0, 0};
    __builtin_amdgcn_tensor_load_to_lds(g0, g1, z4, z4, z8, 0);
#else
    __builtin_amdgcn_tensor_load_to_lds(g0, g1, z4, z4, 0);
#endif
  }
#else
  for (int it = threadIdx.x; it < 64 * 4; it += blockDim.x) {
    const int row = it >> 2;
    const int cg  = (it & 3) << 3;
    *(v8h*)(lds + (size_t)row * LSTRIDE + cg) =
        *(const v8h*)(g + (size_t)row * ld + cg);
  }
#endif
}

// Wait until at most PENDING TDM transfers are outstanding (in-order per
// wave, so PENDING=1 keeps the next tile's DMA in flight). s_wait_tensorcnt
// takes an immediate, so PENDING must be a compile-time constant.
template <int PENDING>
__device__ __forceinline__ void stage_wait() {
#if HAVE_TDM
  if ((threadIdx.x >> 5) == 0)
    __builtin_amdgcn_s_wait_tensorcnt(PENDING);
#endif
}

// ---------------------------------------------------------------------------
// K1: LIF over input x -> spikes (f16) in two transposed layouts:
//   xs_tr [tb][hw][c]          (column-major B for attn GEMM)
//   xs_cm [tb][n=64][k=6144]   (column-major im2col B for the patch conv)
// ---------------------------------------------------------------------------
__global__ void k1_lif_input(const float* __restrict__ x,
                             _Float16* __restrict__ xs_tr,
                             _Float16* __restrict__ xs_cm) {
  const size_t N = (size_t)B_ * C_ * HW_;
  size_t e = (size_t)blockIdx.x * blockDim.x + threadIdx.x;
  if (e >= N) return;
  const int hw = (int)(e % HW_);
  const int c  = (int)((e / HW_) % C_);
  const int b  = (int)(e / ((size_t)HW_ * C_));
  const int yy = hw >> 5, xx = hw & 31;
  const int k = c * 16 + (yy & 3) * 4 + (xx & 3);
  const int n = (yy >> 2) * HP + (xx >> 2);
  float v = 0.f;
#pragma unroll
  for (int t = 0; t < T_; ++t) {
    float xi = x[(size_t)t * N + e];
    v = TAU_INV * (v + xi);                 // v += (x - v)/TAU, TAU=2
    float s = (v >= 1.0f) ? 1.0f : 0.0f;    // heaviside(v - V_TH)
    v *= (1.0f - s);                        // hard reset
    _Float16 sh = (_Float16)s;
    const size_t tb = (size_t)t * B_ + b;
    xs_tr[(tb * HW_ + hw) * C_ + c] = sh;
    xs_cm[(tb * NPATCH + n) * KC + k] = sh;
  }
}

// ---------------------------------------------------------------------------
// K2: weights f32 -> f16.
// ---------------------------------------------------------------------------
__global__ void k2_cvt_weights(const float* __restrict__ wc,
                               const float* __restrict__ wp,
                               _Float16* __restrict__ wcf,
                               _Float16* __restrict__ wpf) {
  size_t i = (size_t)blockIdx.x * blockDim.x + threadIdx.x;
  if (i < (size_t)C2 * KC)  wcf[i] = (_Float16)wc[i];
  if (i < (size_t)C_ * C_)  wpf[i] = (_Float16)wp[i];
}

// ---------------------------------------------------------------------------
// K3: patch-embedding GEMM [768 x 6144] x [6144 x 64] per image + BN1.
// Block = 256 threads (8 waves, M=128); B tiles double-buffered through TDM.
// Output split: (head, d<32) -> y1 [tbh][p][d] ; (head, d>=32) -> y2 [tbh][d][p].
// grid(6, 32).
// ---------------------------------------------------------------------------
__global__ void k3_patch_gemm_bn(const _Float16* __restrict__ Aw,
                                 const _Float16* __restrict__ Bcm,
                                 const float* __restrict__ g1,
                                 const float* __restrict__ b1,
                                 const float* __restrict__ m1,
                                 const float* __restrict__ v1,
                                 _Float16* __restrict__ y1,
                                 _Float16* __restrict__ y2) {
  __shared__ _Float16 lB[2][64 * LSTRIDE];            // 2 x 6 KB ping-pong
  const int tb    = blockIdx.y;
  const int wave  = threadIdx.x >> 5;
  const int strip = blockIdx.x * 8 + wave;            // 0..47, 16 out-chans each
  const _Float16* A = Aw + (size_t)strip * 16 * KC;
  const _Float16* B = Bcm + (size_t)tb * NPATCH * KC; // [64][6144] col-major
  v8f acc[4] = {};
  stage_issue(B, KC, lB[0]);                          // prologue DMA
  int buf = 0;
  for (int k = 0; k < KC; k += 32, buf ^= 1) {
    const bool more = (k + 32 < KC);
    if (more) {
      stage_issue(B + k + 32, KC, lB[buf ^ 1]);       // next tile in flight
      stage_wait<1>();                                // current tile resident
    } else {
      stage_wait<0>();
    }
    __syncthreads();
    __builtin_prefetch(A + k + 32, 0, 1);             // next A chunk
    v16h a = load_a_frag(A + k, KC);
    const _Float16* cur = lB[buf];
#pragma unroll
    for (int j = 0; j < 4; ++j) {
      v16h b = load_bt_frag(cur + j * 16 * LSTRIDE, LSTRIDE, 0);
      acc[j] = wmma16x16x32(a, b, acc[j]);
    }
    __syncthreads();
  }
  const int lane = threadIdx.x & 31;
  const int n0   = lane & 15;
  const int mhi  = (lane >> 4) << 3;
#pragma unroll
  for (int j = 0; j < 4; ++j) {
#pragma unroll
    for (int r = 0; r < 8; ++r) {
      const int o   = strip * 16 + r + mhi;           // out channel 0..767
      const float inv = g1[o] * rsqrtf(v1[o] + EPSV);
      const float bia = b1[o] - m1[o] * inv;
      const float val = acc[j][r] * inv + bia;
      const int n    = j * 16 + n0;                   // patch 0..63
      const int head = o >> 6;
      const int dd   = o & 63;
      const size_t tbh = (size_t)tb * NH + head;
      if (dd < CH) y1[(tbh * NPATCH + n) * CH + dd] = (_Float16)val;       // [p][d]
      else         y2[(tbh * CH + (dd - CH)) * NPATCH + n] = (_Float16)val; // [d][p]
    }
  }
}

// ---------------------------------------------------------------------------
// K4: attn = (y1^T x xs) * scale1 ; LIF over T (register state).
// Per (b,head): GEMM M=64(p), K=32(d), N=1024; B fragments loaded directly
// from column-major xs_tr (no LDS, no barriers).
// Stores spikes column-major: attn_cm [tbh][n=1024][p=64].
// grid(16 n-tiles, 96 bh), block 128 (4 waves = 4 p-tiles).
// ---------------------------------------------------------------------------
__global__ void k4_attn_lif(const _Float16* __restrict__ y1,
                            const _Float16* __restrict__ xs_tr,
                            const float* __restrict__ fr_x,
                            _Float16* __restrict__ attn_cm) {
  const int nt   = blockIdx.x;
  const int bh   = blockIdx.y;
  const int wave = threadIdx.x >> 5;                  // p-tile 0..3
  const int head = bh % NH;
  const int b    = bh / NH;
  const float scale = rsqrtf(fr_x[head] * (float)CH); // 1/sqrt(fr_x * Ch)
  const int lane = threadIdx.x & 31;
  const int n0   = lane & 15;
  const int mhi  = (lane >> 4) << 3;
  v8f vst[4] = {};                                     // LIF membrane state
  for (int t = 0; t < T_; ++t) {
    const size_t tbi = (size_t)t * B_ + b;
    const size_t tbh = tbi * NH + head;
    v16h a = load_a_frag(y1 + (tbh * NPATCH + wave * 16) * CH, CH);
    const _Float16* Bt = xs_tr + (tbi * HW_ + nt * 64) * C_ + head * CH;
#pragma unroll
    for (int j = 0; j < 4; ++j) {
      v16h bf = load_bt_frag(Bt + (size_t)j * 16 * C_, C_, 0);
      v8f acc = {};
      acc = wmma16x16x32(a, bf, acc);
#pragma unroll
      for (int r = 0; r < 8; ++r) {
        const float xi = acc[r] * scale;
        const float nv = TAU_INV * (vst[j][r] + xi);
        const float s  = (nv >= 1.0f) ? 1.0f : 0.0f;
        vst[j][r] = nv * (1.0f - s);
        const int p = wave * 16 + r + mhi;
        const int n = nt * 64 + j * 16 + n0;
        attn_cm[(tbh * HW_ + n) * NPATCH + p] = (_Float16)s;
      }
    }
  }
}

// ---------------------------------------------------------------------------
// K5: out = (y2 x attn) * scale2 ; LIF over T -> spikes column-major
// spk_cm [tb][hw][c]. Per (b,head): M=32(d), K=64(p), N=1024, direct B loads.
// grid(16, 96), block 64 (2 waves = 2 d-tiles).
// ---------------------------------------------------------------------------
__global__ void k5_out_lif(const _Float16* __restrict__ y2,
                           const _Float16* __restrict__ attn_cm,
                           const float* __restrict__ fr_attn,
                           _Float16* __restrict__ spk_cm) {
  const int nt   = blockIdx.x;
  const int bh   = blockIdx.y;
  const int wave = threadIdx.x >> 5;                  // 0..1
  const int head = bh % NH;
  const int b    = bh / NH;
  const float scale = rsqrtf(fr_attn[head] * 64.0f);  // 1/sqrt(fr_attn*LENTH)
  const int lane = threadIdx.x & 31;
  const int n0   = lane & 15;
  const int mhi  = (lane >> 4) << 3;
  v8f vst[4] = {};
  for (int t = 0; t < T_; ++t) {
    const size_t tbi = (size_t)t * B_ + b;
    const size_t tbh = tbi * NH + head;
    const _Float16* A = y2 + (tbh * CH + wave * 16) * NPATCH;
    v16h a0 = load_a_frag(A, NPATCH);                 // K = p 0..31
    v16h a1 = load_a_frag(A + 32, NPATCH);            // K = p 32..63
    const _Float16* Bt = attn_cm + (tbh * HW_ + nt * 64) * NPATCH;
#pragma unroll
    for (int j = 0; j < 4; ++j) {
      const _Float16* Bj = Bt + (size_t)j * 16 * NPATCH;
      v16h b0 = load_bt_frag(Bj, NPATCH, 0);
      v16h b1 = load_bt_frag(Bj, NPATCH, 32);
      v8f acc = {};
      acc = wmma16x16x32(a0, b0, acc);
      acc = wmma16x16x32(a1, b1, acc);
#pragma unroll
      for (int r = 0; r < 8; ++r) {
        const float xi = acc[r] * scale;
        const float nv = TAU_INV * (vst[j][r] + xi);
        const float s  = (nv >= 1.0f) ? 1.0f : 0.0f;
        vst[j][r] = nv * (1.0f - s);
        const int c = head * CH + wave * 16 + r + mhi;
        const int n = nt * 64 + j * 16 + n0;
        spk_cm[(tbi * HW_ + n) * C_ + c] = (_Float16)s;
      }
    }
  }
}

// ---------------------------------------------------------------------------
// K6: projection GEMM [384x384] x [384x1024] per (t,b) + BN2 + residual (f32).
// Block = 256 threads (8 waves, M=128); B tiles double-buffered through TDM
// from column-major spk_cm. grid(16, 3, 32).
// ---------------------------------------------------------------------------
__global__ void k6_proj_bn_res(const _Float16* __restrict__ Wp,
                               const _Float16* __restrict__ Scm,
                               const float* __restrict__ xres,
                               const float* __restrict__ g2,
                               const float* __restrict__ b2,
                               const float* __restrict__ m2,
                               const float* __restrict__ v2,
                               float* __restrict__ out) {
  __shared__ _Float16 lB[2][64 * LSTRIDE];
  const int nt    = blockIdx.x;
  const int wave  = threadIdx.x >> 5;
  const int strip = blockIdx.y * 8 + wave;            // 0..23
  const int tb    = blockIdx.z;
  const _Float16* A = Wp + (size_t)strip * 16 * C_;
  const _Float16* B = Scm + ((size_t)tb * HW_ + nt * 64) * C_;  // [64][384] cm
  v8f acc[4] = {};
  stage_issue(B, C_, lB[0]);
  int buf = 0;
  for (int k = 0; k < C_; k += 32, buf ^= 1) {
    const bool more = (k + 32 < C_);
    if (more) {
      stage_issue(B + k + 32, C_, lB[buf ^ 1]);
      stage_wait<1>();
    } else {
      stage_wait<0>();
    }
    __syncthreads();
    v16h a = load_a_frag(A + k, C_);
    const _Float16* cur = lB[buf];
#pragma unroll
    for (int j = 0; j < 4; ++j) {
      v16h b = load_bt_frag(cur + j * 16 * LSTRIDE, LSTRIDE, 0);
      acc[j] = wmma16x16x32(a, b, acc[j]);
    }
    __syncthreads();
  }
  const int lane = threadIdx.x & 31;
  const int n0   = lane & 15;
  const int mhi  = (lane >> 4) << 3;
#pragma unroll
  for (int j = 0; j < 4; ++j) {
#pragma unroll
    for (int r = 0; r < 8; ++r) {
      const int o = strip * 16 + r + mhi;
      const float inv = g2[o] * rsqrtf(v2[o] + EPSV);
      const float bia = b2[o] - m2[o] * inv;
      const size_t idx = ((size_t)tb * C_ + o) * HW_ + nt * 64 + j * 16 + n0;
      out[idx] = acc[j][r] * inv + bia + xres[idx];
    }
  }
}

// ---------------------------------------------------------------------------
extern "C" void kernel_launch(void* const* d_in, const int* in_sizes, int n_in,
                              void* d_out, int out_size, void* d_ws, size_t ws_size,
                              hipStream_t stream) {
  (void)in_sizes; (void)n_in; (void)out_size; (void)ws_size;
  const float* x       = (const float*)d_in[0];
  const float* w_conv  = (const float*)d_in[1];
  const float* bn1_g   = (const float*)d_in[2];
  const float* bn1_b   = (const float*)d_in[3];
  const float* bn1_m   = (const float*)d_in[4];
  const float* bn1_v   = (const float*)d_in[5];
  const float* w_proj  = (const float*)d_in[6];
  const float* bn2_g   = (const float*)d_in[7];
  const float* bn2_b   = (const float*)d_in[8];
  const float* bn2_m   = (const float*)d_in[9];
  const float* bn2_v   = (const float*)d_in[10];
  const float* fr_x    = (const float*)d_in[11];
  const float* fr_attn = (const float*)d_in[12];
  float* out = (float*)d_out;

  char* ws = (char*)d_ws;
  size_t off = 0;
  auto carve = [&](size_t bytes) -> char* {
    char* p = ws + off;
    off = (off + bytes + 255) & ~(size_t)255;
    return p;
  };
  const size_t NSPK = (size_t)T_ * B_ * C_ * HW_;             // 12.58M elements
  _Float16* xs_tr  = (_Float16*)carve(NSPK * 2);              // 24 MB [tb][hw][c]
  _Float16* xs_cm  = (_Float16*)carve(NSPK * 2);              // 24 MB [tb][n][k]
  _Float16* wconvh = (_Float16*)carve((size_t)C2 * KC * 2);   // 9 MB
  _Float16* wprojh = (_Float16*)carve((size_t)C_ * C_ * 2);   // 0.3 MB
  _Float16* y1buf  = (_Float16*)carve((size_t)T_ * B_ * NH * NPATCH * CH * 2);
  _Float16* y2buf  = (_Float16*)carve((size_t)T_ * B_ * NH * CH * NPATCH * 2);
  _Float16* attncm = (_Float16*)carve((size_t)T_ * B_ * NH * HW_ * NPATCH * 2); // 48 MB
  _Float16* spkcm  = (_Float16*)carve(NSPK * 2);              // 24 MB

  // K1: LIF on input, dual transposed-layout spike emission
  {
    const size_t N = (size_t)B_ * C_ * HW_;
    dim3 g((unsigned)((N + 255) / 256));
    k1_lif_input<<<g, 256, 0, stream>>>(x, xs_tr, xs_cm);
  }
  // K2: weight conversion to f16
  {
    const size_t N = (size_t)C2 * KC;
    dim3 g((unsigned)((N + 255) / 256));
    k2_cvt_weights<<<g, 256, 0, stream>>>(w_conv, w_proj, wconvh, wprojh);
  }
  // K3: patch-embed GEMM + BN1 -> y1 / y2 (double-buffered TDM B)
  k3_patch_gemm_bn<<<dim3(6, 32), 256, 0, stream>>>(
      wconvh, xs_cm, bn1_g, bn1_b, bn1_m, bn1_v, y1buf, y2buf);
  // K4: attn GEMM + scale + LIF -> spikes [n][p]
  k4_attn_lif<<<dim3(16, 96), 128, 0, stream>>>(y1buf, xs_tr, fr_x, attncm);
  // K5: out GEMM + scale + LIF -> spikes [hw][c]
  k5_out_lif<<<dim3(16, 96), 64, 0, stream>>>(y2buf, attncm, fr_attn, spkcm);
  // K6: projection GEMM + BN2 + residual (double-buffered TDM B)
  k6_proj_bn_res<<<dim3(16, 3, 32), 256, 0, stream>>>(
      wprojh, spkcm, x, bn2_g, bn2_b, bn2_m, bn2_v, out);
}
